// EncoderRNN_37005438222496
// MI455X (gfx1250) — matrix operand order repaired and checked by reading, not compiled, not run on hardware
//
#include <hip/hip_runtime.h>
#include <hip/hip_bf16.h>
#include <stdint.h>

// ---------------------------------------------------------------------------
// EncoderRNN (4-layer residual LSTM, packed sequence) for MI455X / gfx1250.
//
// bf16 WMMA GEMMs with a gate-interleaved N permutation (n' = 4*j + gate) so
// each 16x16 WMMA tile holds full (i,f,g,o) quadruples -> the LSTM cell fuses
// into the GEMM wave via lane shuffles. Weights are pre-packed once into the
// exact ISA 7.12.2 B-fragment layout (64 MB bf16, resident in the 192 MB L2);
// activations are packed into the A-fragment layout by the producing kernel so
// each wave loads fragments with two B128 loads per operand. Weight-stream
// prefetch uses inline asm global_prefetch_b8 with scope 0 (WGP) — the only
// scope that pulls into all cache levels (SE/DEV/SYS scopes stop at GL2,
// where the weights already live).
// ---------------------------------------------------------------------------

typedef __bf16 v16bf __attribute__((ext_vector_type(16)));
typedef float  v8f   __attribute__((ext_vector_type(8)));
typedef unsigned int v8u __attribute__((ext_vector_type(8)));

#define HN   1024
#define LN   4
#define BN   64
#define SN   512
#define NT   256              // n' tiles: 4096 / 16
#define KST  32               // k-steps per 1024: 1024 / 32
#define FRAG 512              // halves per fragment: 32 lanes * 16
#define WMAT_HALVES  (NT * KST * FRAG)   // per (layer, matrix): 4,194,304
#define PACKA_HALVES (4 * KST * FRAG)    // per packed activation: 65,536

__device__ __forceinline__ unsigned short f2bf(float f) {
    unsigned int u = __builtin_bit_cast(unsigned int, f);
    u += 0x7FFFu + ((u >> 16) & 1u);     // round to nearest even
    return (unsigned short)(u >> 16);
}

__device__ __forceinline__ v16bf ldfrag(const unsigned short* p) {
    return __builtin_bit_cast(v16bf, *(const v8u*)p);  // two B128 loads
}

// WGP-scope prefetch (scope field 0 = default, omitted in asm): pulls the
// cacheline into all cache levels on miss, stops on WGP hit.
__device__ __forceinline__ void prefetch_wgp(const void* p) {
    asm volatile("global_prefetch_b8 %0, off" :: "v"(p));
}

__device__ __forceinline__ float sigf(float x) {
    return 1.0f / (1.0f + __expf(-x));
}

// A-fragment position for (row m, k K) per ISA 16-bit A 16x32 layout:
// lanes 0-15: K 0-7 -> i=K, K 16-23 -> i=K-8 ; lanes 16-31: K 8-15 -> i=K-8,
// K 24-31 -> i=K-16.
__device__ __forceinline__ void a_pos(int m, int K, int& lp, int& ip) {
    int hi = (K >> 3) & 1;
    lp = m + (hi ? 16 : 0);
    ip = (K & 7) + ((K >= 16) ? 8 : 0);
}

// --------------------------- one-time pack kernels -------------------------

// Pack w_ih / w_hh [L][4096][1024] fp32 into bf16 B-fragments with the
// gate-interleaved column permutation n' = 4*j + gate.
// B layout (32x16, 16-bit): lanes 0-15 hold K 0-7,16-23, lanes 16-31 hold
// K 8-15,24-31 (same order as A); column n' = ntile*16 + (lane & 15).
__global__ void pack_w_kernel(const float* __restrict__ w_ih,
                              const float* __restrict__ w_hh,
                              unsigned short* __restrict__ packW) {
    unsigned int idx = blockIdx.x * blockDim.x + threadIdx.x; // 2^25 total
    int i     = idx & 15;
    int lane  = (idx >> 4) & 31;
    int kst   = (idx >> 9) & 31;
    int ntile = (idx >> 14) & 255;
    int mat   = (idx >> 22) & 1;
    int l     = (int)(idx >> 23);
    if (l >= LN) return;
    int nn = lane & 15;
    int K  = kst * 32 + ((lane >> 4) ? 8 : 0) + (i & 7) + ((i >= 8) ? 16 : 0);
    int np = ntile * 16 + nn;
    int j = np >> 2, gate = np & 3;
    int n = gate * HN + j;                      // original row in [4H]
    const float* w = mat ? w_hh : w_ih;
    float v = w[(l * 4 * HN + n) * HN + K];
    packW[idx] = f2bf(v);
}

// Embedding lookup + pack x_t into A-fragment layout for all timesteps.
__global__ void pack_x_kernel(const int* __restrict__ tok,
                              const float* __restrict__ emb,
                              unsigned short* __restrict__ packX) {
    unsigned int idx = blockIdx.x * blockDim.x + threadIdx.x; // 2^25 total
    int j = idx & (HN - 1);
    int b = (idx >> 10) & 63;
    int t = (int)(idx >> 16);
    if (t >= SN) return;
    int tk = tok[b * SN + t];
    float v = (tk == 0) ? 0.0f : emb[tk * HN + j];   // padding_idx=0 row is 0
    int rt = b >> 4, m = b & 15;
    int kst = j >> 5, K = j & 31;
    int lp, ip; a_pos(m, K, lp, ip);
    packX[(size_t)t * PACKA_HALVES + (rt * KST + kst) * FRAG + lp * 16 + ip] = f2bf(v);
}

__global__ void bias_kernel(const float* __restrict__ b_ih,
                            const float* __restrict__ b_hh,
                            float* __restrict__ biasP) {
    int idx = blockIdx.x * blockDim.x + threadIdx.x;  // 4 * 4096
    if (idx >= LN * 4 * HN) return;
    int np = idx & 4095;
    int l  = idx >> 12;
    int j = np >> 2, gate = np & 3;
    int n = gate * HN + j;
    biasP[idx] = b_ih[l * 4 * HN + n] + b_hh[l * 4 * HN + n];
}

__global__ void zero_kernel(unsigned int* __restrict__ p, unsigned int n) {
    unsigned int i = blockIdx.x * blockDim.x + threadIdx.x;
    if (i < n) p[i] = 0u;
}

// --------------------------- fused layer-step kernel -----------------------
// Grid: 256 blocks (= n' tiles) x 128 threads (4 waves; wave = 16-row batch
// tile). Each wave: gates tile = bias + y@Wih' + h@Whh' via 64 bf16 WMMAs
// (two independent accumulator chains for ILP), then fused LSTM cell via lane
// shuffles (each group of 4 lanes holds i,f,g,o for one (b, j)), masked state
// update, bf16 repack of h / residual y into A-fragment layout.
__global__ __launch_bounds__(128) void lstm_step_kernel(
    const unsigned short* __restrict__ Ay,     // packed y input  [4][32][512]
    const unsigned short* __restrict__ Ah,     // packed h input  [4][32][512]
    const unsigned short* __restrict__ Wih,    // [256][32][512]
    const unsigned short* __restrict__ Whh,
    const float* __restrict__ biasP,           // [4096] in n' order
    float* __restrict__ hbuf,                  // [64][1024] fp32 state
    float* __restrict__ cbuf,
    unsigned short* __restrict__ Ah_out,       // packed h output (other parity)
    unsigned short* __restrict__ Ay_out,       // packed y for next layer
    const int* __restrict__ in_lengths, int t)
{
    const int lane  = threadIdx.x & 31;
    const int rt    = threadIdx.x >> 5;   // batch row-tile 0..3
    const int ntile = blockIdx.x;         // n' tile 0..255

    // C/D layout: VGPR r holds N = lane&15, M = r + 8*(lane>>4).
    float bv = biasP[ntile * 16 + (lane & 15)];
    v8f acc0, acc1;
#pragma unroll
    for (int r = 0; r < 8; ++r) { acc0[r] = bv; acc1[r] = 0.0f; }

    const unsigned short* ay = Ay  + (rt    * KST) * FRAG + lane * 16;
    const unsigned short* ah = Ah  + (rt    * KST) * FRAG + lane * 16;
    const unsigned short* wi = Wih + (ntile * KST) * FRAG + lane * 16;
    const unsigned short* wh = Whh + (ntile * KST) * FRAG + lane * 16;

#pragma unroll 4
    for (int kk = 0; kk < KST; ++kk) {
        // Stage the next k-steps' weight fragments from GL2 into the near
        // cache (WGP scope) ahead of the load clauses.
        prefetch_wgp(wi + (kk + 4) * FRAG);
        prefetch_wgp(wh + (kk + 4) * FRAG);
        v16bf a0 = ldfrag(ay + kk * FRAG);
        v16bf b0 = ldfrag(wi + kk * FRAG);
        acc0 = __builtin_amdgcn_wmma_f32_16x16x32_bf16(
                   false, a0, false, b0, (short)0, acc0, false, false);
        v16bf a1 = ldfrag(ah + kk * FRAG);
        v16bf b1 = ldfrag(wh + kk * FRAG);
        acc1 = __builtin_amdgcn_wmma_f32_16x16x32_bf16(
                   false, a1, false, b1, (short)0, acc1, false, false);
    }

    // Fused LSTM cell. n' = ntile*16 + (lane&15); gate = n'&3; j = n'>>2.
    const int bs   = in_lengths[t];
    const int base = lane & ~3;
    const int j    = ntile * 4 + ((lane & 15) >> 2);
    const int mofs = (lane >> 4) << 3;

#pragma unroll
    for (int r = 0; r < 8; ++r) {
        float v  = acc0[r] + acc1[r];
        float iv = __shfl(v, base + 0, 32);
        float fv = __shfl(v, base + 1, 32);
        float gv = __shfl(v, base + 2, 32);
        float ov = __shfl(v, base + 3, 32);
        int m = r + mofs;
        int b = rt * 16 + m;
        if ((lane & 3) == 0 && b < bs) {      // one writer per (b, j); mask
            int idx = b * HN + j;
            float c_old = cbuf[idx];
            float cn = sigf(fv) * c_old + sigf(iv) * tanhf(gv);
            float hn = sigf(ov) * tanhf(cn);
            cbuf[idx] = cn;
            float h_old = hbuf[idx];
            hbuf[idx] = hn;
            int kst = j >> 5, K = j & 31;
            int lp, ip; a_pos(m, K, lp, ip);
            int p = (rt * KST + kst) * FRAG + lp * 16 + ip;
            Ah_out[p] = f2bf(hn);             // h for next timestep's GEMM
            Ay_out[p] = f2bf(hn + h_old);     // residual feed to next layer
        }
    }
}

__global__ void copy_out_kernel(const float* __restrict__ h3,
                                float* __restrict__ out) {
    int i = blockIdx.x * blockDim.x + threadIdx.x;
    if (i < BN * HN) out[i] = h3[i];
}

// --------------------------------- host ------------------------------------

extern "C" void kernel_launch(void* const* d_in, const int* in_sizes, int n_in,
                              void* d_out, int out_size, void* d_ws, size_t ws_size,
                              hipStream_t stream) {
    const int*   padded_in  = (const int*)d_in[0];   // [64][512]
    const int*   in_lengths = (const int*)d_in[1];   // [512]
    const float* emb        = (const float*)d_in[2]; // [29][1024]
    const float* w_ih       = (const float*)d_in[3]; // [4][4096][1024]
    const float* w_hh       = (const float*)d_in[4];
    const float* b_ih       = (const float*)d_in[5]; // [4][4096]
    const float* b_hh       = (const float*)d_in[6];

    // Workspace layout (~138 MB total)
    char* ws = (char*)d_ws;
    size_t o = 0;
    unsigned short* packW = (unsigned short*)(ws + o); o += (size_t)LN * 2 * WMAT_HALVES * 2; // 64 MB
    unsigned short* packX = (unsigned short*)(ws + o); o += (size_t)SN * PACKA_HALVES * 2;    // 64 MB
    size_t state_begin = o;
    unsigned short* packH = (unsigned short*)(ws + o); o += (size_t)2 * LN * PACKA_HALVES * 2; // 1 MB (parity ping-pong)
    unsigned short* packY = (unsigned short*)(ws + o); o += (size_t)2 * PACKA_HALVES * 2;      // 256 KB
    float* hbuf  = (float*)(ws + o); o += (size_t)LN * BN * HN * 4;                            // 1 MB
    float* cbuf  = (float*)(ws + o); o += (size_t)LN * BN * HN * 4;                            // 1 MB
    float* biasP = (float*)(ws + o); o += (size_t)LN * 4 * HN * 4;                             // 64 KB
    size_t state_bytes = o - state_begin;

    // One-time prep (re-run every call for determinism).
    pack_w_kernel<<<(LN * 2 * WMAT_HALVES + 255) / 256, 256, 0, stream>>>(w_ih, w_hh, packW);
    pack_x_kernel<<<(SN * PACKA_HALVES + 255) / 256, 256, 0, stream>>>(padded_in, emb, packX);
    bias_kernel<<<(LN * 4 * HN + 255) / 256, 256, 0, stream>>>(b_ih, b_hh, biasP);
    unsigned int zwords = (unsigned int)((state_bytes - (size_t)LN * 4 * HN * 4) / 4); // h0=c0=0, packed h=0
    zero_kernel<<<(zwords + 255) / 256, 256, 0, stream>>>((unsigned int*)(ws + state_begin), zwords);

    // Sequential recurrence: one fused GEMM+cell launch per (t, layer).
    for (int t = 0; t < SN; ++t) {
        for (int l = 0; l < LN; ++l) {
            const unsigned short* Ay = (l == 0)
                ? (packX + (size_t)t * PACKA_HALVES)
                : (packY + (size_t)(l & 1) * PACKA_HALVES);
            const unsigned short* Ah  = packH + ((size_t)(t & 1) * LN + l) * PACKA_HALVES;
            unsigned short*       AhO = packH + ((size_t)((t + 1) & 1) * LN + l) * PACKA_HALVES;
            unsigned short*       AyO = packY + (size_t)((l + 1) & 1) * PACKA_HALVES;
            const unsigned short* Wih = packW + (size_t)l * 2 * WMAT_HALVES;
            const unsigned short* Whh = Wih + WMAT_HALVES;
            lstm_step_kernel<<<NT, 128, 0, stream>>>(
                Ay, Ah, Wih, Whh,
                biasP + l * 4 * HN,
                hbuf + (size_t)l * BN * HN,
                cbuf + (size_t)l * BN * HN,
                AhO, AyO, in_lengths, t);
        }
    }

    copy_out_kernel<<<(BN * HN + 255) / 256, 256, 0, stream>>>(
        hbuf + (size_t)3 * BN * HN, (float*)d_out);
}
